// AttentionDecoder_80049600462876
// MI455X (gfx1250) — matrix operand order repaired and checked by reading, not compiled, not run on hardware
//
#include <hip/hip_runtime.h>
#include <hip/hip_bf16.h>
#include <stdint.h>

// Problem constants (match reference)
#define Bdim 48
#define Tdim 1024
#define Ddim 512
#define Hdim 512
#define Fdim 256
#define Cdim 4367
#define STEPS 22

typedef __attribute__((ext_vector_type(16))) __bf16 v16bf;
typedef __attribute__((ext_vector_type(8)))  float  v8f;
typedef unsigned int u32x4 __attribute__((ext_vector_type(4)));
typedef int          i32x8 __attribute__((ext_vector_type(8)));
typedef int          i32x4 __attribute__((ext_vector_type(4)));

// ---------- helpers ----------
__device__ __forceinline__ unsigned short f2bf(float f) {
    unsigned int u = __float_as_uint(f);
    unsigned int r = u + 0x7FFFu + ((u >> 16) & 1u);   // round-to-nearest-even
    return (unsigned short)(r >> 16);
}

__device__ __forceinline__ v16bf make_frag(uint4 lo, uint4 hi) {
    union { unsigned int u[8]; v16bf v; } t;
    t.u[0] = lo.x; t.u[1] = lo.y; t.u[2] = lo.z; t.u[3] = lo.w;
    t.u[4] = hi.x; t.u[5] = hi.y; t.u[6] = hi.z; t.u[7] = hi.w;
    return t.v;
}

__device__ __forceinline__ float sigmoidf(float x) { return 1.0f / (1.0f + __expf(-x)); }

// ---------- one-time prep ----------
__global__ void k_cvt_bf16(const float* __restrict__ x, unsigned short* __restrict__ o, int n4) {
    int i = blockIdx.x * blockDim.x + threadIdx.x;
    if (i < n4) {
        float4 f = ((const float4*)x)[i];
        ushort4 u; u.x = f2bf(f.x); u.y = f2bf(f.y); u.z = f2bf(f.z); u.w = f2bf(f.w);
        ((ushort4*)o)[i] = u;
    }
}

// Wt[f*D + k] = bf16(attn_w1[k*F + f])   (top D rows of attn_w1, transposed to [N,K])
__global__ void k_transpose_w1x(const float* __restrict__ w1, unsigned short* __restrict__ wt) {
    int idx = blockIdx.x * blockDim.x + threadIdx.x;      // F*D total
    int f = idx / Ddim, k = idx % Ddim;
    wt[idx] = f2bf(w1[k * Fdim + f]);
}

__global__ void k_zero(float* __restrict__ p, int n) {
    int i = blockIdx.x * blockDim.x + threadIdx.x;
    if (i < n) p[i] = 0.0f;
}

// ---------- main WMMA GEMM: XW[M=49152, N=256] = Xbf[M,K=512] @ Wt[N,K]^T ----------
// block = 256 threads = 8 waves; wave w -> 16-row tile (blockIdx.x*8 + w), 64 cols (blockIdx.y*64)
// B panel (64 rows x 512 K bf16 = 64KB) staged into LDS once per block via the Tensor Data Mover.
__global__ void __launch_bounds__(256)
k_xw_wmma(const unsigned short* __restrict__ Xb, const unsigned short* __restrict__ Wt,
          float* __restrict__ XW) {
    const int K = Ddim, N = Fdim;
    const int wave = threadIdx.x >> 5;
    const int lane = threadIdx.x & 31;
    const int rowTile = blockIdx.x * 8 + wave;            // 16-row tile
    const int colBase = blockIdx.y * 64;
    const int m = (rowTile << 4) + (lane & 15);           // A row for this lane
    const int kh = lane >> 4;                             // K-half select

    __shared__ unsigned short Bt[64 * 512];               // 64KB B panel

#if __has_builtin(__builtin_amdgcn_tensor_load_to_lds) && __has_builtin(__builtin_amdgcn_s_wait_tensorcnt)
    if (wave == 0) {
        // TDM descriptor (see CDNA5 ISA 8.3/8.4): 2D tile, 64 rows x 512 elems, 2B/elem.
        unsigned long long ga = (unsigned long long)(uintptr_t)(Wt + (size_t)colBase * K);
        unsigned int la = (unsigned int)(uintptr_t)(&Bt[0]);   // LDS byte address = low 32 bits
        u32x4 g0;
        g0[0] = 1u;                                            // count=1, no gather
        g0[1] = la;                                            // lds_addr
        g0[2] = (unsigned int)ga;                              // global_addr[31:0]
        g0[3] = (unsigned int)((ga >> 32) & 0x1FFFFFFu) | (2u << 30);  // addr[56:32] | type=2
        i32x8 g1;
        g1[0] = (int)(1u << 16);                               // data_size=1 (2 bytes)
        g1[1] = (int)(512u << 16);                             // tensor_dim0[15:0]=512 in bits 63:48
        g1[2] = (int)(64u << 16);                              // tensor_dim0 hi=0 | tensor_dim1=64
        g1[3] = (int)(512u << 16);                             // tensor_dim1 hi=0 | tile_dim0=512
        g1[4] = 64;                                            // tile_dim1=64 | tile_dim2=0
        g1[5] = 512;                                           // tensor_dim0_stride[31:0]=512
        g1[6] = 0;                                             // stride0 hi | tensor_dim1_stride lo
        g1[7] = 0;
        i32x4 g2 = {0, 0, 0, 0};
        i32x4 g3 = {0, 0, 0, 0};
        i32x8 g4 = {0, 0, 0, 0, 0, 0, 0, 0};
        __builtin_amdgcn_tensor_load_to_lds(g0, g1, g2, g3, g4, 0);
        __builtin_amdgcn_s_wait_tensorcnt(0);
    }
#else
    // Fallback: cooperative copy (Wt rows colBase..colBase+63 are contiguous)
    {
        const uint4* src = (const uint4*)(Wt + (size_t)colBase * K);
        uint4* dst = (uint4*)Bt;
        for (int i = threadIdx.x; i < 64 * 512 / 8; i += 256) dst[i] = src[i];
    }
#endif
    __syncthreads();

    v8f acc[4] = {};
    const unsigned short* arow = Xb + (size_t)m * K;

    for (int k0 = 0; k0 < K; k0 += 32) {
        // A fragment (16x32 bf16): VGPR0-3 <- K=kh*8+0..7, VGPR4-7 <- K=16+kh*8+0..7
        const uint4* pa = (const uint4*)(arow + k0 + kh * 8);
        uint4 a_lo = pa[0];
        uint4 a_hi = pa[2];                               // +16 bf16 elements
        v16bf A = make_frag(a_lo, a_hi);
#pragma unroll
        for (int c = 0; c < 4; ++c) {
            // B fragment (32x16 bf16) from LDS: lane col = lane&15, K = kh*8+0..7, 16+kh*8+0..7
            const uint4* pb = (const uint4*)(Bt + (c * 16 + (lane & 15)) * 512 + k0 + kh * 8);
            v16bf Bf = make_frag(pb[0], pb[2]);
            acc[c] = __builtin_amdgcn_wmma_f32_16x16x32_bf16(
                false, A, false, Bf, (short)0, acc[c], false, false);
        }
    }
    // D layout: col = lane&15, row = r + 8*(lane>>4)
    const int col0 = colBase + (lane & 15);
#pragma unroll
    for (int c = 0; c < 4; ++c) {
#pragma unroll
        for (int r = 0; r < 8; ++r) {
            int row = (rowTile << 4) + (kh << 3) + r;
            XW[(size_t)row * N + col0 + c * 16] = acc[c][r];
        }
    }
}

// ---------- per-step kernels ----------
// hw[b,f] = b1[f] + sum_k h[b,k] * attn_w1[(D+k)*F + f]
__global__ void k_hw(const float* __restrict__ h, const float* __restrict__ w1,
                     const float* __restrict__ b1, float* __restrict__ hw) {
    int idx = blockIdx.x * blockDim.x + threadIdx.x;      // B*F
    int b = idx / Fdim, f = idx % Fdim;
    float acc = b1[f];
    const float* w = w1 + (size_t)Ddim * Fdim + f;
    const float* hb = h + b * Hdim;
    for (int k = 0; k < Hdim; ++k) acc += hb[k] * w[(size_t)k * Fdim];
    hw[idx] = acc;
}

// scores[bt] = leaky_relu(XW[bt,:] + hw[b,:]) . w2 + b2   (one wave per row)
__global__ void __launch_bounds__(256)
k_score(const float* __restrict__ XW, const float* __restrict__ hw,
        const float* __restrict__ w2, const float* __restrict__ b2,
        float* __restrict__ sc) {
    int wave = threadIdx.x >> 5;
    int lane = threadIdx.x & 31;
    int bt = blockIdx.x * 8 + wave;                       // < B*T
    int b = bt >> 10;                                     // T == 1024
    int base = lane * 8;
    const float4* px = (const float4*)(XW + (size_t)bt * Fdim + base);
    const float4* ph = (const float4*)(hw + b * Fdim + base);
    const float4* pw = (const float4*)(w2 + base);
    float s = 0.0f;
#pragma unroll
    for (int j = 0; j < 2; ++j) {
        float4 xv = px[j], hv = ph[j], wv = pw[j];
        float v0 = xv.x + hv.x, v1 = xv.y + hv.y, v2 = xv.z + hv.z, v3 = xv.w + hv.w;
        v0 = v0 > 0.f ? v0 : 0.01f * v0;
        v1 = v1 > 0.f ? v1 : 0.01f * v1;
        v2 = v2 > 0.f ? v2 : 0.01f * v2;
        v3 = v3 > 0.f ? v3 : 0.01f * v3;
        s += v0 * wv.x + v1 * wv.y + v2 * wv.z + v3 * wv.w;
    }
#pragma unroll
    for (int o = 16; o > 0; o >>= 1) s += __shfl_xor(s, o, 32);
    if (lane == 0) sc[bt] = s + b2[0];
}

// softmax over T per batch
__global__ void __launch_bounds__(256)
k_softmax(const float* __restrict__ sc, float* __restrict__ w) {
    int b = blockIdx.x, tid = threadIdx.x;
    __shared__ float red[256];
    float m = -1e30f;
    for (int t = tid; t < Tdim; t += 256) m = fmaxf(m, sc[b * Tdim + t]);
    red[tid] = m; __syncthreads();
    for (int o = 128; o > 0; o >>= 1) { if (tid < o) red[tid] = fmaxf(red[tid], red[tid + o]); __syncthreads(); }
    m = red[0]; __syncthreads();
    float s = 0.0f;
    for (int t = tid; t < Tdim; t += 256) s += __expf(sc[b * Tdim + t] - m);
    red[tid] = s; __syncthreads();
    for (int o = 128; o > 0; o >>= 1) { if (tid < o) red[tid] += red[tid + o]; __syncthreads(); }
    float inv = 1.0f / red[0];
    for (int t = tid; t < Tdim; t += 256) w[b * Tdim + t] = __expf(sc[b * Tdim + t] - m) * inv;
}

// ctx[b,d] = sum_t x[b,t,d] * w[b,t]     grid: (D/128, B), block 128
__global__ void __launch_bounds__(128)
k_ctx(const float* __restrict__ x, const float* __restrict__ w, float* __restrict__ ctx) {
    int b = blockIdx.y;
    int d = blockIdx.x * 128 + threadIdx.x;
    const float* xb = x + (size_t)b * Tdim * Ddim + d;
    const float* wb = w + b * Tdim;
    float acc = 0.0f;
    for (int t = 0; t < Tdim; ++t) {
        __builtin_prefetch(xb + (size_t)(t + 8) * Ddim, 0, 1);
        acc += xb[(size_t)t * Ddim] * wb[t];
    }
    ctx[b * Ddim + d] = acc;
}

// GRU cell: one thread per (b, j)
__global__ void __launch_bounds__(256)
k_gru(const float* __restrict__ ctx, const float* __restrict__ h,
      const float* __restrict__ wi, const float* __restrict__ wh,
      const float* __restrict__ bi, const float* __restrict__ bh,
      float* __restrict__ hn_out) {
    int idx = blockIdx.x * blockDim.x + threadIdx.x;      // B*H
    int b = idx / Hdim, j = idx % Hdim;
    const float4* c4 = (const float4*)(ctx + b * Ddim);
    const float4* h4 = (const float4*)(h + b * Hdim);
    const float4* wr = (const float4*)(wi + (size_t)j * Ddim);
    const float4* wz = (const float4*)(wi + (size_t)(j + Hdim) * Ddim);
    const float4* wn = (const float4*)(wi + (size_t)(j + 2 * Hdim) * Ddim);
    const float4* vr = (const float4*)(wh + (size_t)j * Hdim);
    const float4* vz = (const float4*)(wh + (size_t)(j + Hdim) * Hdim);
    const float4* vn = (const float4*)(wh + (size_t)(j + 2 * Hdim) * Hdim);
    float ir = bi[j], iz = bi[Hdim + j], in = bi[2 * Hdim + j];
    float hr = bh[j], hz = bh[Hdim + j], hnv = bh[2 * Hdim + j];
    for (int k = 0; k < Ddim / 4; ++k) {
        float4 cv = c4[k], hv = h4[k];
        float4 a = wr[k]; ir += cv.x * a.x + cv.y * a.y + cv.z * a.z + cv.w * a.w;
        float4 bq = wz[k]; iz += cv.x * bq.x + cv.y * bq.y + cv.z * bq.z + cv.w * bq.w;
        float4 cq = wn[k]; in += cv.x * cq.x + cv.y * cq.y + cv.z * cq.z + cv.w * cq.w;
        float4 d = vr[k]; hr += hv.x * d.x + hv.y * d.y + hv.z * d.z + hv.w * d.w;
        float4 e = vz[k]; hz += hv.x * e.x + hv.y * e.y + hv.z * e.z + hv.w * e.w;
        float4 f = vn[k]; hnv += hv.x * f.x + hv.y * f.y + hv.z * f.z + hv.w * f.w;
    }
    float r = sigmoidf(ir + hr);
    float z = sigmoidf(iz + hz);
    float n = tanhf(in + r * hnv);
    hn_out[idx] = (1.0f - z) * n + z * h[idx];
}

// out[b, step, c] = h[b,:] . cls_w[c,:] + cls_b[c]
__global__ void __launch_bounds__(256)
k_cls(const float* __restrict__ h, const float* __restrict__ cw,
      const float* __restrict__ cb, float* __restrict__ out, int step) {
    int idx = blockIdx.x * blockDim.x + threadIdx.x;
    if (idx >= Bdim * Cdim) return;
    int b = idx / Cdim, c = idx % Cdim;
    const float4* h4 = (const float4*)(h + b * Hdim);
    const float4* w4 = (const float4*)(cw + (size_t)c * Hdim);
    float acc = cb[c];
    for (int k = 0; k < Hdim / 4; ++k) {
        float4 hv = h4[k], wv = w4[k];
        acc += hv.x * wv.x + hv.y * wv.y + hv.z * wv.z + hv.w * wv.w;
    }
    out[((size_t)b * STEPS + step) * Cdim + c] = acc;
}

// ---------- launcher ----------
extern "C" void kernel_launch(void* const* d_in, const int* in_sizes, int n_in,
                              void* d_out, int out_size, void* d_ws, size_t ws_size,
                              hipStream_t stream) {
    const float* x   = (const float*)d_in[0];
    const float* aw1 = (const float*)d_in[1];
    const float* ab1 = (const float*)d_in[2];
    const float* aw2 = (const float*)d_in[3];
    const float* ab2 = (const float*)d_in[4];
    const float* gwi = (const float*)d_in[5];
    const float* gwh = (const float*)d_in[6];
    const float* gbi = (const float*)d_in[7];
    const float* gbh = (const float*)d_in[8];
    const float* cw  = (const float*)d_in[9];
    const float* cb  = (const float*)d_in[10];
    float* out = (float*)d_out;

    char* ws = (char*)d_ws;
    size_t off = 0;
    auto alloc = [&](size_t bytes) { void* p = ws + off; off = (off + bytes + 255) & ~(size_t)255; return p; };
    unsigned short* Xbf = (unsigned short*)alloc((size_t)Bdim * Tdim * Ddim * 2);
    unsigned short* Wt  = (unsigned short*)alloc((size_t)Fdim * Ddim * 2);
    float* XW  = (float*)alloc((size_t)Bdim * Tdim * Fdim * 4);
    float* hw  = (float*)alloc((size_t)Bdim * Fdim * 4);
    float* sc  = (float*)alloc((size_t)Bdim * Tdim * 4);
    float* wsm = (float*)alloc((size_t)Bdim * Tdim * 4);
    float* ctx = (float*)alloc((size_t)Bdim * Ddim * 4);
    float* hA  = (float*)alloc((size_t)Bdim * Hdim * 4);
    float* hB  = (float*)alloc((size_t)Bdim * Hdim * 4);

    // One-time prep
    {
        int n4 = Bdim * Tdim * Ddim / 4;
        k_cvt_bf16<<<n4 / 256, 256, 0, stream>>>(x, Xbf, n4);
        k_transpose_w1x<<<(Fdim * Ddim) / 256, 256, 0, stream>>>(aw1, Wt);
        k_zero<<<(Bdim * Hdim) / 256, 256, 0, stream>>>(hA, Bdim * Hdim);
        dim3 g((Bdim * Tdim / 16) / 8, Fdim / 64);
        k_xw_wmma<<<g, 256, 0, stream>>>(Xbf, Wt, XW);
    }

    float* h_cur = hA;
    float* h_nxt = hB;
    for (int t = 0; t < STEPS; ++t) {
        k_hw<<<(Bdim * Fdim) / 256, 256, 0, stream>>>(h_cur, aw1, ab1, hw);
        k_score<<<(Bdim * Tdim) / 8, 256, 0, stream>>>(XW, hw, aw2, ab2, sc);
        k_softmax<<<Bdim, 256, 0, stream>>>(sc, wsm);
        k_ctx<<<dim3(Ddim / 128, Bdim), 128, 0, stream>>>(x, wsm, ctx);
        k_gru<<<(Bdim * Hdim) / 256, 256, 0, stream>>>(ctx, h_cur, gwi, gwh, gbi, gbh, h_nxt);
        k_cls<<<(Bdim * Cdim + 255) / 256, 256, 0, stream>>>(h_nxt, cw, cb, out, t);
        float* tmp = h_cur; h_cur = h_nxt; h_nxt = tmp;
    }
}
